// CSNN_9165460210320
// MI455X (gfx1250) — compile-verified, matrix operand order checked by reading, not run on hardware
//
#include <hip/hip_runtime.h>

// ---------------- CDNA5 WMMA types ----------------
typedef __attribute__((ext_vector_type(16))) _Float16 v16h;
typedef __attribute__((ext_vector_type(8)))  _Float16 v8h;
typedef __attribute__((ext_vector_type(8)))  float    v8f;

// Problem constants
#define BATCH 16
#define TSTEPS 15

// ---------------------------------------------------------------------------
// Weight packing: layout wp[((nt*KC + kc)*32 + lane)*16 + e]
// lane&15 = N column within 16-wide tile, g = lane>>4 selects K-subset:
//   e < 8  -> k = kc*32 + 8*g + e
//   e >= 8 -> k = kc*32 + 8*g + e + 8      (i.e. {0..7,16..23} / {8..15,24..31})
// Mirrors the ISA 16-bit fragment lane/K striping so the fragment load is one
// contiguous 32B global load per lane.
// ---------------------------------------------------------------------------
__global__ void pack_w_kernel(const float* __restrict__ w, _Float16* __restrict__ wp,
                              int COUT, int K, int NT, int KC) {
    int idx = blockIdx.x * blockDim.x + threadIdx.x;
    int total = NT * KC * 32 * 16;
    if (idx >= total) return;
    int e    = idx & 15;
    int lane = (idx >> 4) & 31;
    int rest = idx >> 9;
    int kc   = rest % KC;
    int nt   = rest / KC;
    int g    = lane >> 4;
    int n    = nt * 16 + (lane & 15);
    int k    = kc * 32 + 8 * g + e + (e >= 8 ? 8 : 0);
    float val = (n < COUT && k < K) ? w[(long)n * K + k] : 0.0f;
    wp[idx] = (_Float16)val;
}

// ---------------------------------------------------------------------------
// Implicit-GEMM conv via v_wmma_f32_16x16x32_f16.
// M = flattened (b,h,w) (w fastest; W % 16 == 0 so a 16-row M tile is 16
// consecutive w in one (b,h) row). One wave per block; A tile staged in LDS
// and the single A fragment is reused across all NT output-channel tiles.
//
// Staging: lane L owns K-index k = kc*32+L; the (ci,kh,kw) decode, row bounds
// test and base address are computed ONCE per (lane, kc); the 16 M-row gathers
// are then contiguous global addresses (per-element cost: 1 cmp + load + ds).
// Single-wave workgroup => LDS ordering needs only s_wait_dscnt, no barrier.
// ---------------------------------------------------------------------------
template<int CIN, int COUT, int KH, int KW, int H, int W, int PAD, int NT, bool IN_F32>
__global__ __launch_bounds__(32)
void conv_wmma_kernel(const void* __restrict__ in, long in_b_stride,
                      const _Float16* __restrict__ wpack, float* __restrict__ u) {
    constexpr int K  = CIN * KH * KW;
    constexpr int KC = (K + 31) / 32;

    const int lane = threadIdx.x;          // 0..31
    const int m0   = blockIdx.x * 16;      // 16 consecutive w positions
    const int b    = m0 / (H * W);
    const int rem  = m0 % (H * W);
    const int h    = rem / W;
    const int w0   = rem % W;

    __shared__ __align__(16) _Float16 atile[16][32];

    v8f acc[NT];
#pragma unroll
    for (int nt = 0; nt < NT; ++nt) { v8f z = {}; acc[nt] = z; }

    const v16h* __restrict__ wpv = reinterpret_cast<const v16h*>(wpack);

    // A-fragment addressing (constant per lane)
    const int arow  = lane & 15;
    const int abase = (lane >> 4) * 8;

    for (int kc = 0; kc < KC; ++kc) {
        // ---- stage one 16x32 A tile: lane owns one K column ----
        const int k = kc * 32 + lane;
        int ci = k / (KH * KW);
        int r2 = k % (KH * KW);
        int kh = r2 / KW;
        int kw = r2 % KW;
        int ih = h + kh - PAD;
        const bool rowvalid = (k < K) & (ih >= 0) & (ih < H);
        const long base = (long)b * in_b_stride + (long)ci * H * W + (long)ih * W;
        const int  iwb  = w0 + kw - PAD;
#pragma unroll
        for (int r = 0; r < 16; ++r) {
            int iw = iwb + r;
            _Float16 val = (_Float16)0.0f;
            if (rowvalid && (unsigned)iw < (unsigned)W) {
                if (IN_F32) val = (_Float16)((const float*)in)[base + iw];
                else        val = ((const _Float16*)in)[base + iw];
            }
            atile[r][lane] = val;
        }
        // Single wave: DS pipe is in-order per wave; wait + compiler fence.
        asm volatile("s_wait_dscnt 0" ::: "memory");

        // A fragment: lane<16 holds K {0..7,16..23}, lane>=16 holds {8..15,24..31}
        union { v16h v; v8h h8[2]; } af;
        af.h8[0] = *(const v8h*)&atile[arow][abase];
        af.h8[1] = *(const v8h*)&atile[arow][abase + 16];
        asm volatile("" ::: "memory");   // keep reads before next iter's stores

#pragma unroll
        for (int nt = 0; nt < NT; ++nt) {
            v16h bf = wpv[((long)nt * KC + kc) * 32 + lane];
            acc[nt] = __builtin_amdgcn_wmma_f32_16x16x32_f16(
                false, af.v, false, bf, (short)0, acc[nt], false, false);
        }
    }

    // Epilogue: C layout — VGPR v, lanes 0-15: M=v, lanes 16-31: M=8+v; N=lane&15
    const int ncol = lane & 15;
    const int mg   = lane >> 4;
#pragma unroll
    for (int nt = 0; nt < NT; ++nt) {
        int n = nt * 16 + ncol;
        if (n < COUT) {
            float* __restrict__ up = u + (((long)b * COUT + n) * H + h) * W + (w0 + 8 * mg);
#pragma unroll
            for (int v = 0; v < 8; ++v) up[v] = acc[nt][v];
        }
    }
}

// ---------------------------------------------------------------------------
// Fused membrane-update + fire + 2x2 maxpool. One thread per pooled output.
// "fired/inactive" is encoded as pot = -inf (never re-fires, no act mask).
// ---------------------------------------------------------------------------
template<bool ACCUM_OUT>
__global__ void fire_pool_kernel(const float* __restrict__ u, float* __restrict__ pot,
                                 void* __restrict__ sout, float thresh,
                                 int C, int H, int W) {
    const int Hp = H >> 1, Wp = W >> 1;
    long tid = (long)blockIdx.x * blockDim.x + threadIdx.x;
    long total = (long)BATCH * C * Hp * Wp;
    if (tid >= total) return;
    int j = tid % Wp;
    int i = (tid / Wp) % Hp;
    long bc = tid / ((long)Wp * Hp);   // b*C + c

    bool any = false;
#pragma unroll
    for (int dy = 0; dy < 2; ++dy) {
#pragma unroll
        for (int dx = 0; dx < 2; ++dx) {
            long idx = (bc * H + (2 * i + dy)) * W + (2 * j + dx);
            float p = pot[idx] + u[idx];
            bool fired = p > thresh;
            pot[idx] = fired ? -__builtin_inff() : p;
            any |= fired;
        }
    }
    if (ACCUM_OUT) {
        ((float*)sout)[tid] += any ? 1.0f : 0.0f;
    } else {
        ((_Float16*)sout)[tid] = any ? (_Float16)1.0f : (_Float16)0.0f;
    }
}

// ---------------------------------------------------------------------------
extern "C" void kernel_launch(void* const* d_in, const int* in_sizes, int n_in,
                              void* d_out, int out_size, void* d_ws, size_t ws_size,
                              hipStream_t stream) {
    (void)in_sizes; (void)n_in;
    const float* x  = (const float*)d_in[0];
    const float* w1 = (const float*)d_in[1];
    const float* w2 = (const float*)d_in[2];
    const float* w3 = (const float*)d_in[3];

    // ---- workspace layout (256B aligned regions) ----
    auto align256 = [](size_t v) { return (v + 255) & ~(size_t)255; };
    const size_t POT1 = align256((size_t)BATCH * 30  * 128 * 128 * 4);
    const size_t POT2 = align256((size_t)BATCH * 100 * 64  * 64  * 4);
    const size_t POT3 = align256((size_t)BATCH * 200 * 32  * 32  * 4);
    const size_t UBUF = POT1;                                   // largest u
    const size_t S1   = align256((size_t)BATCH * 30  * 64 * 64 * 2);
    const size_t S2   = align256((size_t)BATCH * 100 * 32 * 32 * 2);
    const size_t WP1  = align256((size_t)2  * 3  * 32 * 16 * 2);
    const size_t WP2  = align256((size_t)7  * 9  * 32 * 16 * 2);
    const size_t WP3  = align256((size_t)13 * 29 * 32 * 16 * 2);

    char* p = (char*)d_ws;
    float*    pot1 = (float*)p;        p += POT1;
    float*    pot2 = (float*)p;        p += POT2;
    float*    pot3 = (float*)p;        p += POT3;
    float*    u    = (float*)p;        p += UBUF;
    _Float16* s1   = (_Float16*)p;     p += S1;
    _Float16* s2   = (_Float16*)p;     p += S2;
    _Float16* wp1  = (_Float16*)p;     p += WP1;
    _Float16* wp2  = (_Float16*)p;     p += WP2;
    _Float16* wp3  = (_Float16*)p;     p += WP3;
    if ((size_t)(p - (char*)d_ws) > ws_size) return;  // insufficient scratch

    // ---- deterministic per-call init ----
    hipMemsetAsync(pot1, 0, (size_t)BATCH * 30  * 128 * 128 * 4, stream);
    hipMemsetAsync(pot2, 0, (size_t)BATCH * 100 * 64  * 64  * 4, stream);
    hipMemsetAsync(pot3, 0, (size_t)BATCH * 200 * 32  * 32  * 4, stream);
    hipMemsetAsync(d_out, 0, (size_t)out_size * 4, stream);

    // ---- pack weights into WMMA fragment layout ----
    {
        int t1 = 2 * 3 * 512, t2 = 7 * 9 * 512, t3 = 13 * 29 * 512;
        pack_w_kernel<<<(t1 + 255) / 256, 256, 0, stream>>>(w1, wp1, 30, 75,  2,  3);
        pack_w_kernel<<<(t2 + 255) / 256, 256, 0, stream>>>(w2, wp2, 100, 270, 7,  9);
        pack_w_kernel<<<(t3 + 255) / 256, 256, 0, stream>>>(w3, wp3, 200, 900, 13, 29);
    }

    const long XB_STRIDE = (long)TSTEPS * 3 * 128 * 128;   // per-batch stride of x
    const long XT_STRIDE = (long)3 * 128 * 128;            // per-timestep offset

    const int MT1 = BATCH * 128 * 128 / 16;   // 16384 wave-tiles
    const int MT2 = BATCH * 64  * 64  / 16;   // 4096
    const int MT3 = BATCH * 32  * 32  / 16;   // 1024

    const long FP1 = (long)BATCH * 30  * 64 * 64;
    const long FP2 = (long)BATCH * 100 * 32 * 32;
    const long FP3 = (long)BATCH * 200 * 16 * 16;

    for (int t = 0; t < TSTEPS; ++t) {
        // layer 1: x(f32) -> u -> fire/pool -> s1
        conv_wmma_kernel<3, 30, 5, 5, 128, 128, 2, 2, true>
            <<<MT1, 32, 0, stream>>>((const void*)(x + (long)t * XT_STRIDE),
                                     XB_STRIDE, wp1, u);
        fire_pool_kernel<false><<<(int)((FP1 + 255) / 256), 256, 0, stream>>>(
            u, pot1, (void*)s1, 5.0f, 30, 128, 128);

        // layer 2: s1(f16) -> u -> fire/pool -> s2
        conv_wmma_kernel<30, 100, 3, 3, 64, 64, 1, 7, false>
            <<<MT2, 32, 0, stream>>>((const void*)s1, (long)30 * 64 * 64, wp2, u);
        fire_pool_kernel<false><<<(int)((FP2 + 255) / 256), 256, 0, stream>>>(
            u, pot2, (void*)s2, 1.0f, 100, 64, 64);

        // layer 3: s2(f16) -> u -> fire/pool -> accumulate into d_out
        conv_wmma_kernel<100, 200, 3, 3, 32, 32, 1, 13, false>
            <<<MT3, 32, 0, stream>>>((const void*)s2, (long)100 * 32 * 32, wp3, u);
        fire_pool_kernel<true><<<(int)((FP3 + 255) / 256), 256, 0, stream>>>(
            u, pot3, d_out, 1.0f, 200, 32, 32);
    }
}
